// DistPredictor_79164837200023
// MI455X (gfx1250) — matrix operand order repaired and checked by reading, not compiled, not run on hardware
//
#include <hip/hip_runtime.h>
#include <hip/hip_bf16.h>

typedef __attribute__((ext_vector_type(16))) __bf16 v16bf;
typedef __attribute__((ext_vector_type(8)))  float  v8f;
typedef __attribute__((ext_vector_type(4)))  float  v4f;

#define N_NODES 100000
#define N_EDGES 600000
#define DIM 128

// ---------------------------------------------------------------------------
// Prep: build bf16 B-operands for both GEMMs + the fused second bias.
//   Bw[n][k] = W_w[n][k]                      (B for hp = h @ W_w^T)
//   G = W^T H;  BG[n][k] = G[k][n] = sum_j W_w[j][k] * H[j][n]
//   bH[n] = sum_k W_b[k] * H[k][n]            (bias of hpH = h@G + b@H)
// 16384 threads, one (n,k) each. Trivial cost (~4 MFLOP).
// ---------------------------------------------------------------------------
__global__ void prep_kernel(const float* __restrict__ W_w,
                            const float* __restrict__ W_b,
                            const float* __restrict__ H,
                            __bf16* __restrict__ Bw,
                            __bf16* __restrict__ BG,
                            float*  __restrict__ bH) {
    int t = blockIdx.x * blockDim.x + threadIdx.x;   // 0..16383
    int n = t >> 7;
    int k = t & 127;
    Bw[t] = (__bf16)W_w[t];                          // t == n*128+k
    float g = 0.f;
    for (int j = 0; j < DIM; ++j)
        g += W_w[j * DIM + k] * H[j * DIM + n];
    BG[t] = (__bf16)g;
    if (t < DIM) {
        float s = 0.f;
        for (int kk = 0; kk < DIM; ++kk)
            s += W_b[kk] * H[kk * DIM + t];
        bH[t] = s;
    }
}

// ---------------------------------------------------------------------------
// Node transform: one wave per 16-row tile of h. Single pass over h computing
// BOTH hp = h@W^T + b and hpH = h@G + bH with shared bf16 A-fragments.
// 16 f32 accumulator tiles (128 VGPRs), 64 v_wmma_f32_16x16x32_bf16 per wave.
// ---------------------------------------------------------------------------
__global__ void __launch_bounds__(256)
node_gemm_kernel(const float*  __restrict__ h,
                 const __bf16* __restrict__ Bw,
                 const __bf16* __restrict__ BG,
                 const float*  __restrict__ W_b,
                 const float*  __restrict__ bH,
                 float* __restrict__ hp,
                 float* __restrict__ hpH,
                 int ntiles) {
    int wid  = (blockIdx.x * blockDim.x + threadIdx.x) >> 5;
    int lane = threadIdx.x & 31;
    if (wid >= ntiles) return;                // wave-uniform: EXEC stays all-1s

    const int rowA = lane & 15;               // A: lane holds row (lane&15)
    const int kgrp = lane >> 4;               // lane group selects K half
    const int colB = lane & 15;               // B: lane holds column (lane&15)

    const float* hrow = h + (long)(wid * 16 + rowA) * DIM;

    v8f acc[16];
    #pragma unroll
    for (int i = 0; i < 16; ++i) acc[i] = (v8f)0.f;

    #pragma unroll
    for (int kt = 0; kt < 4; ++kt) {
        const int kbase = kt * 32;
        // A-fragment: lane holds row rowA, 16 contiguous K values
        //             K=kbase+kgrp*16..+15 (matches B)
        v4f f0 = *(const v4f*)(hrow + kbase + kgrp * 16);
        v4f f1 = *(const v4f*)(hrow + kbase + kgrp * 16 + 4);
        v4f f2 = *(const v4f*)(hrow + kbase + kgrp * 16 + 8);
        v4f f3 = *(const v4f*)(hrow + kbase + kgrp * 16 + 12);
        v16bf A;
        A[0]  = (__bf16)f0.x; A[1]  = (__bf16)f0.y; A[2]  = (__bf16)f0.z; A[3]  = (__bf16)f0.w;
        A[4]  = (__bf16)f1.x; A[5]  = (__bf16)f1.y; A[6]  = (__bf16)f1.z; A[7]  = (__bf16)f1.w;
        A[8]  = (__bf16)f2.x; A[9]  = (__bf16)f2.y; A[10] = (__bf16)f2.z; A[11] = (__bf16)f2.w;
        A[12] = (__bf16)f3.x; A[13] = (__bf16)f3.y; A[14] = (__bf16)f3.z; A[15] = (__bf16)f3.w;

        #pragma unroll
        for (int nt = 0; nt < 8; ++nt) {
            // B-fragment: lane holds column colB, 16 contiguous K values
            const __bf16* bwp = Bw + ((nt * 16 + colB) * DIM + kbase + kgrp * 16);
            v16bf Bf = *(const v16bf*)bwp;
            acc[nt] = __builtin_amdgcn_wmma_f32_16x16x32_bf16(
                false, A, false, Bf, (short)0, acc[nt], false, false);

            const __bf16* bgp = BG + ((nt * 16 + colB) * DIM + kbase + kgrp * 16);
            v16bf Gf = *(const v16bf*)bgp;
            acc[8 + nt] = __builtin_amdgcn_wmma_f32_16x16x32_bf16(
                false, A, false, Gf, (short)0, acc[8 + nt], false, false);
        }
    }

    // bias + store (C layout: VGPR r -> M = r + 8*kgrp, lane -> N = nt*16+colB)
    const int mbase = wid * 16 + kgrp * 8;
    #pragma unroll
    for (int nt = 0; nt < 8; ++nt) {
        int col  = nt * 16 + colB;
        float b1 = W_b[col];
        float b2 = bH[col];
        #pragma unroll
        for (int r = 0; r < 8; ++r) {
            hp [(long)(mbase + r) * DIM + col] = acc[nt][r]     + b1;
            hpH[(long)(mbase + r) * DIM + col] = acc[8 + nt][r] + b2;
        }
    }
}

// ---------------------------------------------------------------------------
// Edge scores: one wave per edge. 4 f32 per lane (coalesced b128 row loads,
// hp+hpH are 102 MB total -> L2 resident), squared diff, wave32 shfl reduce.
// ---------------------------------------------------------------------------
__global__ void __launch_bounds__(256)
edge_score_kernel(const float* __restrict__ hp,
                  const float* __restrict__ hpH,
                  const int*   __restrict__ src,
                  const int*   __restrict__ dst,
                  float* __restrict__ out, int nedges) {
    int wid  = (blockIdx.x * blockDim.x + threadIdx.x) >> 5;
    int lane = threadIdx.x & 31;
    if (wid >= nedges) return;                // wave-uniform guard

    long s = (long)src[wid];
    long d = (long)dst[wid];
    v4f a = *(const v4f*)(hp  + s * DIM + lane * 4);
    v4f b = *(const v4f*)(hpH + d * DIM + lane * 4);
    v4f df = a - b;
    float sum = df.x * df.x + df.y * df.y + df.z * df.z + df.w * df.w;
    #pragma unroll
    for (int off = 16; off > 0; off >>= 1)
        sum += __shfl_xor(sum, off, 32);
    if (lane == 0) out[wid] = sum;
}

// ---------------------------------------------------------------------------
extern "C" void kernel_launch(void* const* d_in, const int* in_sizes, int n_in,
                              void* d_out, int out_size, void* d_ws, size_t ws_size,
                              hipStream_t stream) {
    const float* h   = (const float*)d_in[0];
    const int*   src = (const int*)  d_in[1];
    const int*   dst = (const int*)  d_in[2];
    const float* W_w = (const float*)d_in[3];
    const float* W_b = (const float*)d_in[4];
    const float* H   = (const float*)d_in[5];
    float* out = (float*)d_out;

    // Workspace layout (all offsets 256B aligned):
    //   hp  : N*128 f32  (51.2 MB)
    //   hpH : N*128 f32  (51.2 MB)   -> hp+hpH fit in 192 MB L2 for gathers
    //   Bw  : 128*128 bf16 (32 KB)
    //   BG  : 128*128 bf16 (32 KB)
    //   bH  : 128 f32
    char*   ws  = (char*)d_ws;
    float*  hp  = (float*) (ws);
    float*  hpH = (float*) (ws + (size_t)N_NODES * DIM * 4);
    __bf16* Bw  = (__bf16*)(ws + (size_t)N_NODES * DIM * 8);
    __bf16* BG  = (__bf16*)(ws + (size_t)N_NODES * DIM * 8 + 32768);
    float*  bH  = (float*) (ws + (size_t)N_NODES * DIM * 8 + 65536);

    prep_kernel<<<64, 256, 0, stream>>>(W_w, W_b, H, Bw, BG, bH);

    const int ntiles = N_NODES / 16;                 // 6250 (exact)
    int gblocks = (ntiles * 32 + 255) / 256;
    node_gemm_kernel<<<gblocks, 256, 0, stream>>>(h, Bw, BG, W_b, bH, hp, hpH, ntiles);

    int eblocks = ((long)N_EDGES * 32 + 255) / 256;  // 8 edges (waves) per block
    edge_score_kernel<<<eblocks, 256, 0, stream>>>(hp, hpH, src, dst, out, N_EDGES);
}
